// PointTransformerSeg_11982958756085
// MI455X (gfx1250) — compile-verified, matrix operand order checked by reading, not run on hardware
//
#include <hip/hip_runtime.h>

// ---------------------------------------------------------------------------
// PointTransformer-seg forward for MI455X (gfx1250, wave32, WMMA).
// All channel-mixing ops are f16 WMMA GEMMs (f32 accumulate) over the
// concatenated column space NT = B*N = 8192. Attention numerators are
// materialized in f16 (67MB, fits MI455X's 192MB L2); row-softmax denom is
// folded into V, the L1 column renorm into the residual-diff kernel.
// Both WMMA operands are staged in LDS in fragment-row-major order so every
// fragment load is two contiguous ds_load_b128 per lane (no scalar gathers).
// ---------------------------------------------------------------------------

typedef __attribute__((ext_vector_type(16))) _Float16 v16h;
typedef __attribute__((ext_vector_type(8)))  float    v8f;

#define NT    8192      // total columns = B * N
#define NPTS  4096      // points per batch
#define BN_EPS 1e-5f
#define FPAD  40        // padded LDS row (halves); bank-conflict-free reads

// ---- WMMA fragment load (CDNA5 ISA 7.12.2 16-bit layouts) -----------------
// Tile stored [frag_row][k] row-major, FPAD stride. lane<16 -> row=lane,
// halves 0..7 => K=e+8*h2 (contig 16B), 8..15 => K=16+8*h2+e (contig 16B).
// Used for A (row=m) and for B staged transposed (row=n).
__device__ __forceinline__ v16h frag_ld(const _Float16* s, int lane) {
  const int row = lane & 15, h2 = lane >> 4;
  const _Float16* p = s + row * FPAD + 8 * h2;
  v16h f;
#pragma unroll
  for (int e = 0; e < 8; ++e) f[e] = p[e];
#pragma unroll
  for (int e = 0; e < 8; ++e) f[8 + e] = p[8 + e + 8];   // base+16+8*h2+e
  return f;
}

// ---- generic WMMA GEMM: O[m,n] = epi( sum_k A[m,k]*B[k,n] ) ---------------
// 256 threads = 8 waves as 4(M)x2(N); block tile 64(M) x 128(N); KT=32.
// Each wave: 16x64 output = 4 wmma per K-step on one A fragment.
// epilogue: +bias[m], BN(g,b) fresh stats, act(0,1=relu,2=leaky.2), +residual.
__global__ __launch_bounds__(256) void k_gemm(
    const _Float16* __restrict__ A, int lda,
    const _Float16* __restrict__ Bm, int ldb,
    _Float16* __restrict__ O, int ldc,
    int M, int K,
    const float* __restrict__ bias,
    const float* __restrict__ bng, const float* __restrict__ bnb,
    int act, const _Float16* __restrict__ resid)
{
  __shared__ _Float16 As[64 * FPAD];     // [m][k]
  __shared__ _Float16 BsT[128 * FPAD];   // [n][k] (transposed staging)
  const int tid  = threadIdx.x;
  const int lane = tid & 31, wave = tid >> 5;
  const int wm = wave >> 1, wn = wave & 1;
  const int m0 = blockIdx.y * 64;
  const int n0 = blockIdx.x * 128;
  v8f acc[4] = {{}, {}, {}, {}};
  const int ar = tid >> 2, ac = (tid & 3) * 8;    // A stage: 64 rows x 32
  const int br = tid >> 3, bc = (tid & 7) * 16;   // B stage: 32 k-rows x 128 n

  for (int k0 = 0; k0 < K; k0 += 32) {
    { // stage A [64][32] -> As[m][k], contiguous 16B per thread
      const int gm = m0 + ar;
      _Float16* dst = &As[ar * FPAD + ac];
      if (gm < M) {
        const _Float16* src = A + (size_t)gm * lda + k0 + ac;
#pragma unroll
        for (int i = 0; i < 8; ++i) dst[i] = src[i];
      } else {
#pragma unroll
        for (int i = 0; i < 8; ++i) dst[i] = (_Float16)0.f;
      }
    }
    { // stage B [32][128] transposed -> BsT[n][k]; coalesced 32B global reads
      const _Float16* src = Bm + (size_t)(k0 + br) * ldb + n0 + bc;
#pragma unroll
      for (int i = 0; i < 16; ++i) BsT[(bc + i) * FPAD + br] = src[i];
      if (k0 + 32 < K)  // hint next K-tile toward L2 (global_prefetch_b8)
        __builtin_prefetch(Bm + (size_t)(k0 + 32 + br) * ldb + n0 + bc, 0, 1);
    }
    __syncthreads();
    const v16h af = frag_ld(&As[wm * 16 * FPAD], lane);
#pragma unroll
    for (int t = 0; t < 4; ++t) {
      const v16h bf = frag_ld(&BsT[(wn * 64 + t * 16) * FPAD], lane);
      acc[t] = __builtin_amdgcn_wmma_f32_16x16x32_f16(false, af, false, bf,
                                                      (short)0, acc[t],
                                                      false, false);
    }
    __syncthreads();
  }

  const int lcol = lane & 15, h2 = lane >> 4;
  const float bns = rsqrtf(1.f + BN_EPS);
#pragma unroll
  for (int t = 0; t < 4; ++t) {
    const int n = n0 + wn * 64 + t * 16 + lcol;
#pragma unroll
    for (int r = 0; r < 8; ++r) {
      const int m = m0 + wm * 16 + h2 * 8 + r;
      if (m < M) {
        float v = acc[t][r];
        if (bias) v += bias[m];
        if (bng)  v = v * (bng[m] * bns) + bnb[m];
        if (act == 1)      v = fmaxf(v, 0.f);
        else if (act == 2) v = (v > 0.f) ? v : 0.2f * v;
        if (resid) v += (float)resid[(size_t)m * ldc + n];
        O[(size_t)m * ldc + n] = (_Float16)v;
      }
    }
  }
}

// ---- attention: e = q^T q (K=32), row softmax, store p=exp(e-max) f16 -----
// One block = 16 rows of one batch; 8 waves lockstep over 4096 columns,
// two passes (rowmax, then exp/rowsum/store). rowinv = 1/rowsum.
__global__ __launch_bounds__(256) void k_attn(
    const _Float16* __restrict__ q,   // [32][NT]
    _Float16* __restrict__ att,       // [2][NPTS][NPTS] raw exp numerators
    float* __restrict__ rowinv)       // [2][NPTS]
{
  __shared__ _Float16 Aq[16 * FPAD];     // [n][c]
  __shared__ _Float16 BqT[128 * FPAD];   // [m][c] (transposed staging)
  __shared__ float part[8 * 16];
  __shared__ float redv[16];
  const int tid = threadIdx.x;
  const int lane = tid & 31, wv = tid >> 5;
  const int h2 = lane >> 4;
  const int bb = blockIdx.y;
  const int n0 = blockIdx.x * 16;
  const int cbase = bb * NPTS;

  // stage transposed q tile: Aq[nl][c] = q[c][cbase+n0+nl]
  for (int i = tid; i < 512; i += 256) {
    const int nl = i & 15, c = i >> 4;
    Aq[nl * FPAD + c] = q[(size_t)c * NT + cbase + n0 + nl];
  }
  __syncthreads();
  const v16h af = frag_ld(Aq, lane);

  float rmax[8];
#pragma unroll
  for (int r = 0; r < 8; ++r) rmax[r] = -3.0e38f;

  // pass 1: row max
  for (int it = 0; it < NPTS / 128; ++it) {
    const int mb = it * 128;
    for (int i = tid; i < 32 * 128; i += 256) {
      const int c = i >> 7, ml = i & 127;     // coalesced read, scattered store
      BqT[ml * FPAD + c] = q[(size_t)c * NT + cbase + mb + ml];
    }
    __syncthreads();
    const v16h bf = frag_ld(&BqT[wv * 16 * FPAD], lane);
    v8f z = {};
    v8f e = __builtin_amdgcn_wmma_f32_16x16x32_f16(false, af, false, bf,
                                                   (short)0, z, false, false);
#pragma unroll
    for (int r = 0; r < 8; ++r) rmax[r] = fmaxf(rmax[r], e[r]);
    __syncthreads();
  }
#pragma unroll
  for (int off = 1; off < 16; off <<= 1)
#pragma unroll
    for (int r = 0; r < 8; ++r)
      rmax[r] = fmaxf(rmax[r], __shfl_xor(rmax[r], off, 32));
  if ((lane & 15) == 0) {
#pragma unroll
    for (int r = 0; r < 8; ++r) part[wv * 16 + h2 * 8 + r] = rmax[r];
  }
  __syncthreads();
  if (tid < 16) {
    float m = -3.0e38f;
    for (int w = 0; w < 8; ++w) m = fmaxf(m, part[w * 16 + tid]);
    redv[tid] = m;
  }
  __syncthreads();
  float rm[8], rsum[8];
#pragma unroll
  for (int r = 0; r < 8; ++r) { rm[r] = redv[h2 * 8 + r]; rsum[r] = 0.f; }

  // pass 2: p = exp(e - rowmax); accumulate rowsum; store p (f16)
  for (int it = 0; it < NPTS / 128; ++it) {
    const int mb = it * 128;
    for (int i = tid; i < 32 * 128; i += 256) {
      const int c = i >> 7, ml = i & 127;
      BqT[ml * FPAD + c] = q[(size_t)c * NT + cbase + mb + ml];
    }
    __syncthreads();
    const v16h bf = frag_ld(&BqT[wv * 16 * FPAD], lane);
    v8f z = {};
    v8f e = __builtin_amdgcn_wmma_f32_16x16x32_f16(false, af, false, bf,
                                                   (short)0, z, false, false);
    const int m = mb + wv * 16 + (lane & 15);
#pragma unroll
    for (int r = 0; r < 8; ++r) {
      const float p = __expf(e[r] - rm[r]);
      rsum[r] += p;
      att[((size_t)(bb * NPTS + n0 + h2 * 8 + r)) * NPTS + m] = (_Float16)p;
    }
    __syncthreads();
  }
#pragma unroll
  for (int off = 1; off < 16; off <<= 1)
#pragma unroll
    for (int r = 0; r < 8; ++r) rsum[r] += __shfl_xor(rsum[r], off, 32);
  if ((lane & 15) == 0) {
#pragma unroll
    for (int r = 0; r < 8; ++r) part[wv * 16 + h2 * 8 + r] = rsum[r];
  }
  __syncthreads();
  if (tid < 16) {
    float s = 0.f;
    for (int w = 0; w < 8; ++w) s += part[w * 16 + tid];
    rowinv[bb * NPTS + n0 + tid] = 1.f / s;
  }
}

// ---- small elementwise / reduction kernels --------------------------------
__global__ void k_f2h(const float* __restrict__ s, _Float16* __restrict__ d, int n) {
  int i = blockIdx.x * 256 + threadIdx.x;
  if (i < n) d[i] = (_Float16)s[i];
}

// h1 = relu(bn1(conv1_w @ x^T)) ; K=3 scalar path
__global__ void k_conv1(const float* __restrict__ x, const float* __restrict__ w,
                        const float* __restrict__ g, const float* __restrict__ b,
                        _Float16* __restrict__ h1) {
  int i = blockIdx.x * 256 + threadIdx.x;
  if (i >= 128 * NT) return;
  const int o = i >> 13, j = i & (NT - 1);
  float s = w[o * 3 + 0] * x[(size_t)j * 3 + 0] +
            w[o * 3 + 1] * x[(size_t)j * 3 + 1] +
            w[o * 3 + 2] * x[(size_t)j * 3 + 2];
  s = s * (g[o] * rsqrtf(1.f + BN_EPS)) + b[o];
  h1[i] = (_Float16)fmaxf(s, 0.f);
}

// v_scaled = v * rowinv[col]
__global__ void k_scale_v(const _Float16* __restrict__ v, const float* __restrict__ ri,
                          _Float16* __restrict__ o) {
  int i = blockIdx.x * 256 + threadIdx.x;
  if (i >= 128 * NT) return;
  o[i] = (_Float16)((float)v[i] * ri[i & (NT - 1)]);
}

// colsum[b][m] = sum_n rowinv[b][n] * p[b][n][m]
__global__ void k_colsum(const _Float16* __restrict__ att, const float* __restrict__ ri,
                         float* __restrict__ cs) {
  const int bb = blockIdx.y;
  const int m = blockIdx.x * 256 + threadIdx.x;
  const _Float16* ap = att + (size_t)bb * NPTS * NPTS + m;
  const float* rp = ri + bb * NPTS;
  float s = 0.f;
  for (int n = 0; n < NPTS; ++n) s += rp[n] * (float)ap[(size_t)n * NPTS];
  cs[bb * NPTS + m] = s;
}

// diff = h - xnum / (1e-9 + colsum[col])
__global__ void k_diff(const _Float16* __restrict__ h, const _Float16* __restrict__ xn,
                       const float* __restrict__ cs, _Float16* __restrict__ d) {
  int i = blockIdx.x * 256 + threadIdx.x;
  if (i >= 128 * NT) return;
  const int j = i & (NT - 1);
  d[i] = (_Float16)((float)h[i] - (float)xn[i] / (1e-9f + cs[j]));
}

// per-row max/mean pooling over N for hf rows (M=1024)
__global__ void k_pool(const _Float16* __restrict__ hf, float* __restrict__ xmax,
                       float* __restrict__ xavg) {
  const int o = blockIdx.x, bb = blockIdx.y, tid = threadIdx.x;
  const _Float16* p = hf + (size_t)o * NT + bb * NPTS;
  float mx = -3.0e38f, sm = 0.f;
  for (int i = tid; i < NPTS; i += 256) {
    const float v = (float)p[i];
    mx = fmaxf(mx, v); sm += v;
  }
  __shared__ float s1[256], s2[256];
  s1[tid] = mx; s2[tid] = sm;
  __syncthreads();
  for (int s = 128; s > 0; s >>= 1) {
    if (tid < s) { s1[tid] = fmaxf(s1[tid], s1[tid + s]); s2[tid] += s2[tid + s]; }
    __syncthreads();
  }
  if (tid == 0) {
    xmax[bb * 1024 + o] = s1[0];
    xavg[bb * 1024 + o] = s2[0] * (1.f / NPTS);
  }
}

// cls_f = leaky(bn(label_w @ cls_label))  (K=16, 128 outputs total)
__global__ void k_cls(const float* __restrict__ w, const float* __restrict__ lab,
                      const float* __restrict__ g, const float* __restrict__ b,
                      float* __restrict__ out) {
  const int t = threadIdx.x;
  if (t >= 128) return;
  const int bb = t >> 6, o = t & 63;
  float s = 0.f;
  for (int c = 0; c < 16; ++c) s += w[o * 16 + c] * lab[bb * 16 + c];
  s = s * (g[o] * rsqrtf(1.f + BN_EPS)) + b[o];
  out[t] = (s > 0.f) ? s : 0.2f * s;
}

// broadcast pooled features + cls into big[1024..3135][:]
__global__ void k_bcast(const float* __restrict__ xmax, const float* __restrict__ xavg,
                        const float* __restrict__ clsf, _Float16* __restrict__ big) {
  int i = blockIdx.x * 256 + threadIdx.x;
  if (i >= 2112 * NT) return;
  const int r = i >> 13, j = i & (NT - 1), bb = j >> 12;
  float v;
  if (r < 1024)      v = xmax[bb * 1024 + r];
  else if (r < 2048) v = xavg[bb * 1024 + (r - 1024)];
  else               v = clsf[bb * 64 + (r - 2048)];
  big[(size_t)(1024 + r) * NT + j] = (_Float16)v;
}

// out3 f16 [50][NT] -> d_out f32 [B][50][N]
__global__ void k_repack(const _Float16* __restrict__ o3, float* __restrict__ out) {
  int i = blockIdx.x * 256 + threadIdx.x;
  if (i >= 2 * 50 * NPTS) return;
  const int bb = i / (50 * NPTS);
  const int r = i - bb * 50 * NPTS;
  const int o = r >> 12, n = r & (NPTS - 1);
  out[i] = (float)o3[(size_t)o * NT + bb * NPTS + n];
}

// ---------------------------------------------------------------------------
extern "C" void kernel_launch(void* const* d_in, const int* in_sizes, int n_in,
                              void* d_out, int out_size, void* d_ws, size_t ws_size,
                              hipStream_t stream) {
  (void)in_sizes; (void)n_in; (void)out_size; (void)ws_size;
  const float* x         = (const float*)d_in[0];
  const float* cls_lab   = (const float*)d_in[1];
  const float* conv1_w   = (const float*)d_in[2];
  const float* bn1_g     = (const float*)d_in[3];
  const float* bn1_b     = (const float*)d_in[4];
  const float* conv2_w   = (const float*)d_in[5];
  const float* bn2_g     = (const float*)d_in[6];
  const float* bn2_b     = (const float*)d_in[7];
  const float* sa_qk_w   = (const float*)d_in[8];
  const float* sa_v_w    = (const float*)d_in[9];
  const float* sa_v_b    = (const float*)d_in[10];
  const float* sa_t_w    = (const float*)d_in[11];
  const float* sa_t_b    = (const float*)d_in[12];
  const float* sa_bn_g   = (const float*)d_in[13];
  const float* sa_bn_b   = (const float*)d_in[14];
  const float* fuse_w    = (const float*)d_in[15];
  const float* fuse_bn_g = (const float*)d_in[16];
  const float* fuse_bn_b = (const float*)d_in[17];
  const float* label_w   = (const float*)d_in[18];
  const float* label_bn_g= (const float*)d_in[19];
  const float* label_bn_b= (const float*)d_in[20];
  const float* convs1_w  = (const float*)d_in[21];
  const float* convs1_b  = (const float*)d_in[22];
  const float* bns1_g    = (const float*)d_in[23];
  const float* bns1_b    = (const float*)d_in[24];
  const float* convs2_w  = (const float*)d_in[25];
  const float* convs2_b  = (const float*)d_in[26];
  const float* bns2_g    = (const float*)d_in[27];
  const float* bns2_b    = (const float*)d_in[28];
  const float* convs3_w  = (const float*)d_in[29];
  const float* convs3_b  = (const float*)d_in[30];

  char* wp = (char*)d_ws;
  auto alloc = [&](size_t bytes) -> char* {
    char* p = wp; wp += (bytes + 255) & ~(size_t)255; return p;
  };
  auto h16 = [&](size_t n) { return (_Float16*)alloc(n * 2); };
  auto f32 = [&](size_t n) { return (float*)alloc(n * 4); };

  // f16 weight copies
  _Float16* wc2   = h16(128 * 128);
  _Float16* wqk   = h16(4 * 32 * 128);
  _Float16* wv    = h16(4 * 128 * 128);
  _Float16* wt    = h16(4 * 128 * 128);
  _Float16* wfuse = h16(1024 * 512);
  _Float16* ws1c  = h16(512 * 3136);
  _Float16* ws2c  = h16(256 * 512);
  _Float16* ws3c  = h16(50 * 256);
  // activations
  _Float16* h1    = h16((size_t)128 * NT);
  _Float16* h2b   = h16((size_t)128 * NT);
  _Float16* qbuf  = h16((size_t)32 * NT);
  _Float16* vbuf  = h16((size_t)128 * NT);
  _Float16* vsc   = h16((size_t)128 * NT);
  _Float16* attb  = h16((size_t)2 * NPTS * NPTS);
  _Float16* xnum  = h16((size_t)128 * NT);
  _Float16* dbuf  = h16((size_t)128 * NT);
  _Float16* hcat  = h16((size_t)512 * NT);
  _Float16* big   = h16((size_t)3136 * NT);
  _Float16* hs1   = h16((size_t)512 * NT);
  _Float16* hs2   = h16((size_t)256 * NT);
  _Float16* out3  = h16((size_t)50 * NT);
  float* rowinv = f32(2 * NPTS);
  float* colsum = f32(2 * NPTS);
  float* xmax   = f32(2 * 1024);
  float* xavg   = f32(2 * 1024);
  float* clsf   = f32(2 * 64);

  auto cvt = [&](const float* s, _Float16* d, int n) {
    k_f2h<<<(n + 255) / 256, 256, 0, stream>>>(s, d, n);
  };
  auto gemm = [&](const _Float16* A, int lda, const _Float16* Bm, int ldb,
                  _Float16* O, int ldc, int M, int Ncols, int K,
                  const float* bias, const float* g, const float* be, int act,
                  const _Float16* resid) {
    dim3 grid(Ncols / 128, (M + 63) / 64);
    k_gemm<<<grid, 256, 0, stream>>>(A, lda, Bm, ldb, O, ldc, M, K,
                                     bias, g, be, act, resid);
  };

  // weight conversions
  cvt(conv2_w, wc2, 128 * 128);
  cvt(sa_qk_w, wqk, 4 * 32 * 128);
  cvt(sa_v_w,  wv,  4 * 128 * 128);
  cvt(sa_t_w,  wt,  4 * 128 * 128);
  cvt(fuse_w,  wfuse, 1024 * 512);
  cvt(convs1_w, ws1c, 512 * 3136);
  cvt(convs2_w, ws2c, 256 * 512);
  cvt(convs3_w, ws3c, 50 * 256);

  // stem
  k_conv1<<<(128 * NT + 255) / 256, 256, 0, stream>>>(x, conv1_w, bn1_g, bn1_b, h1);
  gemm(wc2, 128, h1, NT, h2b, NT, 128, NT, 128, nullptr, bn2_g, bn2_b, 1, nullptr);

  // 4 self-attention blocks; feats written directly into hcat rows
  const _Float16* hprev = h2b;
  for (int i = 0; i < 4; ++i) {
    gemm(wqk + (size_t)i * 32 * 128, 128, hprev, NT, qbuf, NT, 32, NT, 128,
         nullptr, nullptr, nullptr, 0, nullptr);
    gemm(wv + (size_t)i * 128 * 128, 128, hprev, NT, vbuf, NT, 128, NT, 128,
         sa_v_b + i * 128, nullptr, nullptr, 0, nullptr);
    k_attn<<<dim3(NPTS / 16, 2), 256, 0, stream>>>(qbuf, attb, rowinv);
    k_scale_v<<<(128 * NT + 255) / 256, 256, 0, stream>>>(vbuf, rowinv, vsc);
    k_colsum<<<dim3(NPTS / 256, 2), 256, 0, stream>>>(attb, rowinv, colsum);
    for (int b = 0; b < 2; ++b)
      gemm(vsc + b * NPTS, NT, attb + (size_t)b * NPTS * NPTS, NPTS,
           xnum + b * NPTS, NT, 128, NPTS, NPTS,
           nullptr, nullptr, nullptr, 0, nullptr);
    k_diff<<<(128 * NT + 255) / 256, 256, 0, stream>>>(hprev, xnum, colsum, dbuf);
    _Float16* hout = hcat + (size_t)i * 128 * NT;
    gemm(wt + (size_t)i * 128 * 128, 128, dbuf, NT, hout, NT, 128, NT, 128,
         sa_t_b + i * 128, sa_bn_g + i * 128, sa_bn_b + i * 128, 1, hprev);
    hprev = hout;
  }

  // fuse (writes rows 0..1023 of big), pooling, cls, broadcast
  gemm(wfuse, 512, hcat, NT, big, NT, 1024, NT, 512,
       nullptr, fuse_bn_g, fuse_bn_b, 2, nullptr);
  k_pool<<<dim3(1024, 2), 256, 0, stream>>>(big, xmax, xavg);
  k_cls<<<1, 128, 0, stream>>>(label_w, cls_lab, label_bn_g, label_bn_b, clsf);
  k_bcast<<<(2112 * NT + 255) / 256, 256, 0, stream>>>(xmax, xavg, clsf, big);

  // segmentation head
  gemm(ws1c, 3136, big, NT, hs1, NT, 512, NT, 3136,
       convs1_b, bns1_g, bns1_b, 1, nullptr);
  gemm(ws2c, 512, hs1, NT, hs2, NT, 256, NT, 512,
       convs2_b, bns2_g, bns2_b, 1, nullptr);
  gemm(ws3c, 256, hs2, NT, out3, NT, 50, NT, 256,
       convs3_b, nullptr, nullptr, 0, nullptr);
  k_repack<<<(2 * 50 * NPTS + 255) / 256, 256, 0, stream>>>(out3, (float*)d_out);
}
